// RateRNN_41008347742413
// MI455X (gfx1250) — matrix operand order repaired and checked by reading, not compiled, not run on hardware
//
#include <hip/hip_runtime.h>
#include <hip/hip_bf16.h>

// ---------------------------------------------------------------------------
// Rate RNN on MI455X (gfx1250, wave32, WMMA + TDM)
//   rec = r @ W.T  via V_WMMA_F32_16X16X32_BF16
//   persistent kernel, W slice resident in LDS (staged by tensor_load_to_lds),
//   grid barrier per step, outputs = rtrace @ Wout done in a final GEMV pass.
// ---------------------------------------------------------------------------

#define NN      2048
#define TSTEPS  1000
#define BATCH   64
#define ALPHA_C 0.9f
#define BETA_C  0.1f
#define NWG     128          // one 16-neuron column tile per workgroup
#define BLOCK   128          // 4 waves: one 16x16 C tile (batch x neuron) each
#define KTILES  64           // 2048 / 32

typedef __attribute__((ext_vector_type(16))) __bf16   v16bf;
typedef __attribute__((ext_vector_type(8)))  __bf16   v8bf;
typedef __attribute__((ext_vector_type(8)))  float    v8f;
typedef __attribute__((ext_vector_type(4)))  float    v4f;
typedef __attribute__((ext_vector_type(4)))  unsigned v4u;
typedef __attribute__((ext_vector_type(8)))  int      v8i;
typedef __attribute__((ext_vector_type(4)))  int      v4i;

// ---------------------------------------------------------------------------
// Pack W = relu(W_raw) * D * M into bf16, arranged per WMMA B-fragment layout:
//   Wb[((nt*64 + kt)*32 + lane)*16 + e] = W[nt*16 + (lane&15)][kt*32 + (lane>>4)*16 + e]
// so in the hot loop a lane reads its whole B fragment as 32 contiguous bytes.
// ---------------------------------------------------------------------------
__global__ __launch_bounds__(256) void pack_w_kernel(
    const float* __restrict__ Wraw, const float* __restrict__ D,
    const float* __restrict__ M, __bf16* __restrict__ Wb) {
  unsigned ridx = blockIdx.x * 256u + threadIdx.x;   // 262144 rows of 16 elems
  unsigned l    = ridx & 31u;
  unsigned kt   = (ridx >> 5) & 63u;
  unsigned nt   = ridx >> 11;
  unsigned j    = nt * 16u + (l & 15u);
  unsigned i0   = kt * 32u + (l >> 4) * 16u;
  const float* wrow = Wraw + (size_t)j * NN + i0;
  const float* mrow = M    + (size_t)j * NN + i0;
  const float* drow = D + i0;
  __bf16* dst = Wb + (size_t)ridx * 16u;
#pragma unroll
  for (int e = 0; e < 16; ++e) {
    float w = wrow[e];
    w = w > 0.f ? w : 0.f;
    w *= drow[e] * mrow[e];
    dst[e] = (__bf16)w;
  }
}

// ---------------------------------------------------------------------------
// Per-launch init: r0 = 0.5 (bf16), reset grid barrier.
// ---------------------------------------------------------------------------
__global__ __launch_bounds__(256) void init_state_kernel(
    __bf16* __restrict__ rbuf, unsigned* __restrict__ bar) {
  unsigned tid = blockIdx.x * 256u + threadIdx.x;    // 131072 threads
  rbuf[tid]              = (__bf16)0.5f;             // r buffer 0 (step 0 input)
  rbuf[BATCH * NN + tid] = (__bf16)0.0f;             // r buffer 1
  if (tid < 2) bar[tid] = 0u;
}

// ---------------------------------------------------------------------------
// Persistent RNN kernel. 128 WGs x 128 threads (4 waves).
//   WG w owns neurons [16w, 16w+16); wave m computes batch rows [16m, 16m+16).
//   x state lives in the WMMA accumulator layout across all 1000 steps.
// ---------------------------------------------------------------------------
__global__ __launch_bounds__(BLOCK) void rnn_persistent_kernel(
    const float* __restrict__ inputs,    // (64, 1000)
    const __bf16* __restrict__ Wb,       // packed B-fragment layout
    const float* __restrict__ Win,       // (2048)
    float* __restrict__ rtrace,          // (64, 1000, 2048)
    __bf16* __restrict__ rbuf,           // 2 x (64, 2048)
    unsigned* __restrict__ bar) {
  __shared__ __bf16 ldsW[KTILES * 32 * 16];          // 64 KB W slice, resident

  const unsigned wg   = blockIdx.x;
  const unsigned tid  = threadIdx.x;
  const unsigned lane = tid & 31u;
  const unsigned wave = tid >> 5;
  const unsigned half = lane >> 4;
  const unsigned l15  = lane & 15u;
  const unsigned n_base = wg * 16u;
  const unsigned m_base = wave * 16u;

  // ---- stage this WG's W slice (16 neurons x 2048 K, bf16 = 64 KB) to LDS ---
#if __has_builtin(__builtin_amdgcn_tensor_load_to_lds)
  if (wave == 0) {
    // D# descriptor: 2D tensor, 8-byte elements, 2048 x 4 tile == 64 KB copy.
    unsigned long long ga =
        (unsigned long long)(const void*)(Wb + (size_t)wg * (KTILES * 32 * 16));
    unsigned lds_off = (unsigned)(unsigned long long)(void*)&ldsW[0]; // addr[31:0]
    v4u g0;
    g0[0] = 1u;                                    // count=1, user descriptor
    g0[1] = lds_off;                               // lds_addr
    g0[2] = (unsigned)(ga & 0xffffffffull);        // global_addr[31:0]
    g0[3] = (unsigned)((ga >> 32) & 0x01ffffffull) // global_addr[56:32]
            | (2u << 30);                          // type = 2 ("image")
    v8i g1;
    g1[0] = (int)(3u << 16);                       // data_size = 8 bytes
    g1[1] = (int)((2048u & 0xffffu) << 16);        // tensor_dim0 = 2048 (lo16)
    g1[2] = (int)(4u << 16);                       // dim0 hi16=0 | tensor_dim1 = 4
    g1[3] = (int)(2048u << 16);                    // dim1 hi16=0 | tile_dim0 = 2048
    g1[4] = 4;                                     // tile_dim1 = 4, tile_dim2 = 0
    g1[5] = 2048;                                  // tensor_dim0_stride lo32
    g1[6] = 0;                                     // stride hi | dim1_stride lo
    g1[7] = 0;
    v4i gz4 = {0, 0, 0, 0};                        // groups 2/3 unused (<=2D)
    v8i gz8 = {0, 0, 0, 0, 0, 0, 0, 0};            // extra group (this toolchain)
    __builtin_amdgcn_tensor_load_to_lds(g0, g1, gz4, gz4, gz8, 0);
    __builtin_amdgcn_s_wait_tensorcnt(0);
  }
#else
  {
    const v4u* src = (const v4u*)(Wb + (size_t)wg * (KTILES * 32 * 16));
    v4u* dst = (v4u*)ldsW;
    for (unsigned i = tid; i < (KTILES * 32 * 16) / 8; i += BLOCK)
      dst[i] = src[i];
  }
#endif
  __syncthreads();

  const unsigned ng = n_base + l15;        // this lane's neuron (fixed)
  const float win = Win[ng];

  v8f x;                                   // persistent x state (C-tile layout)
#pragma unroll
  for (int v = 0; v < 8; ++v) x[v] = 0.0f;

  __bf16* r0 = rbuf;
  __bf16* r1 = rbuf + BATCH * NN;
  const unsigned boff = lane * 16u;        // B fragment offset inside a k-tile

  for (int t = 0; t < TSTEPS; ++t) {
    const __bf16* rc = (t & 1) ? r1 : r0;
    __bf16*       rn = (t & 1) ? r0 : r1;

    // hoist the 8 per-lane input terms so they overlap the WMMA loop
    float uin[8];
#pragma unroll
    for (int v = 0; v < 8; ++v)
      uin[v] = inputs[(int)(m_base + v + 8 * half) * TSTEPS + t] * win;

    // A fragment row for this lane: m = m_base + l15, K = h*16 .. h*16+15
    const __bf16* arow = rc + (size_t)(m_base + l15) * NN + half * 16u;

    v8f rec;
#pragma unroll
    for (int v = 0; v < 8; ++v) rec[v] = 0.0f;

#pragma unroll 4
    for (int kt = 0; kt < KTILES; ++kt) {
      v8bf alo = *(const v8bf*)(arow + kt * 32);        // K = kt*32 + h*16 ..
      v8bf ahi = *(const v8bf*)(arow + kt * 32 + 8);    // K = kt*32+h*16+8 ..
      v16bf a = __builtin_shufflevector(alo, ahi, 0, 1, 2, 3, 4, 5, 6, 7,
                                        8, 9, 10, 11, 12, 13, 14, 15);
      v16bf b = *(const v16bf*)(ldsW + kt * 512 + boff);
      rec = __builtin_amdgcn_wmma_f32_16x16x32_bf16(
          false, a, false, b, (short)0, rec, false, false);
    }

    // ---- state update + trace/state stores (no reduction in hot loop) ----
#pragma unroll
    for (int v = 0; v < 8; ++v) {
      int m = (int)(m_base + v + 8 * half);             // batch row of x[v]
      float xn = ALPHA_C * x[v] + BETA_C * (rec[v] + uin[v]);
      x[v] = xn;
      float r = __builtin_amdgcn_rcpf(1.0f + __expf(-xn));   // sigmoid
      rn[(size_t)m * NN + ng] = (__bf16)r;
      __builtin_nontemporal_store(
          r, &rtrace[((size_t)m * TSTEPS + t) * (size_t)NN + ng]);
    }

    // ---- grid-wide barrier: all r_new visible before next step ----
    __syncthreads();
    if (tid == 0) {
      __threadfence();
      unsigned g = __hip_atomic_load(&bar[1], __ATOMIC_RELAXED,
                                     __HIP_MEMORY_SCOPE_AGENT);
      unsigned a = __hip_atomic_fetch_add(&bar[0], 1u, __ATOMIC_ACQ_REL,
                                          __HIP_MEMORY_SCOPE_AGENT);
      if (a == NWG - 1) {
        __hip_atomic_store(&bar[0], 0u, __ATOMIC_RELAXED,
                           __HIP_MEMORY_SCOPE_AGENT);
        __hip_atomic_store(&bar[1], g + 1u, __ATOMIC_RELEASE,
                           __HIP_MEMORY_SCOPE_AGENT);
      } else {
        while (__hip_atomic_load(&bar[1], __ATOMIC_ACQUIRE,
                                 __HIP_MEMORY_SCOPE_AGENT) == g) {
          __builtin_amdgcn_s_sleep(1);
        }
      }
    }
    __syncthreads();
  }
}

// ---------------------------------------------------------------------------
// Final pass: outputs[b,t] = sum_n rtrace[b,t,n] * Wout[n].
// One wave per (b,t) row; 524 MB streamed at HBM rate, ~23 us.
// ---------------------------------------------------------------------------
__global__ __launch_bounds__(256) void out_gemv_kernel(
    const float* __restrict__ rtrace, const float* __restrict__ Wout,
    float* __restrict__ outputs) {
  unsigned row  = blockIdx.x * 8u + (threadIdx.x >> 5);  // 64000 rows
  unsigned lane = threadIdx.x & 31u;
  const float* rp = rtrace + (size_t)row * NN;
  float s = 0.0f;
#pragma unroll
  for (int i = 0; i < 16; ++i) {                 // 2048 = 16 * 32 lanes * 4
    unsigned off = (i * 32u + lane) * 4u;
    v4f rv = __builtin_nontemporal_load((const v4f*)(rp + off));
    v4f wv = *(const v4f*)(Wout + off);
    s += rv[0] * wv[0] + rv[1] * wv[1] + rv[2] * wv[2] + rv[3] * wv[3];
  }
  s += __shfl_xor(s, 1, 32);
  s += __shfl_xor(s, 2, 32);
  s += __shfl_xor(s, 4, 32);
  s += __shfl_xor(s, 8, 32);
  s += __shfl_xor(s, 16, 32);
  if (lane == 0) outputs[row] = s;
}

// ---------------------------------------------------------------------------
extern "C" void kernel_launch(void* const* d_in, const int* in_sizes, int n_in,
                              void* d_out, int out_size, void* d_ws,
                              size_t ws_size, hipStream_t stream) {
  const float* inputs = (const float*)d_in[0];   // (64,1000,1)
  const float* Wraw   = (const float*)d_in[1];   // (2048,2048)
  const float* Wout   = (const float*)d_in[2];   // (1,2048)
  const float* Win    = (const float*)d_in[3];   // (2048,1)
  const float* D      = (const float*)d_in[4];   // (1,2048)
  const float* M      = (const float*)d_in[5];   // (2048,2048)

  float* outputs = (float*)d_out;                           // 64*1000
  float* rtrace  = (float*)d_out + (size_t)BATCH * TSTEPS;  // 64*1000*2048

  char* ws = (char*)d_ws;
  unsigned* bar = (unsigned*)ws;                                    // 256 B
  __bf16* rbuf  = (__bf16*)(ws + 256);                              // 512 KB
  __bf16* Wb    = (__bf16*)(ws + 256 + (size_t)2 * BATCH * NN * 2); // 8 MB

  pack_w_kernel<<<1024, 256, 0, stream>>>(Wraw, D, M, Wb);
  init_state_kernel<<<512, 256, 0, stream>>>(rbuf, bar);
  rnn_persistent_kernel<<<NWG, BLOCK, 0, stream>>>(
      inputs, Wb, Win, rtrace, rbuf, bar);
  out_gemv_kernel<<<8000, 256, 0, stream>>>(rtrace, Wout, outputs);
}
